// TopKSparseVAttention_22204980920456
// MI455X (gfx1250) — matrix operand order repaired and checked by reading, not compiled, not run on hardware
//
#include <hip/hip_runtime.h>
#include <math.h>

typedef __attribute__((ext_vector_type(16))) __bf16 v16bf;
typedef __attribute__((ext_vector_type(8)))  float  v8f;

namespace {

constexpr int   kSeq    = 4096;
constexpr int   kDim    = 128;        // head dim == d_model (H=1)
constexpr int   kTileM  = 16;         // query rows per block
constexpr int   kTiles  = kSeq / kTileM;  // 256
constexpr int   kWaves  = 8;          // waves per block
constexpr int   kBatch  = 4;
constexpr float kNegInf = -3.0e38f;
constexpr float kLn10k  = 9.210340371976184f;   // ln(10000)
constexpr float kScale  = 0.08838834764831845f; // 1/sqrt(128)

// ---------- RoPE + fragment building ----------

__device__ __forceinline__ void rope8(const float* __restrict__ row, float pos,
                                      int fbase, float* __restrict__ dst) {
#pragma unroll
  for (int j = 0; j < 4; ++j) {
    const int f = fbase + 2 * j;
    const float xe = row[f];
    const float xo = row[f + 1];
    const float ang = pos * __expf(-((float)f * (1.0f / (float)kDim)) * kLn10k);
    float s, c;
    __sincosf(ang, &s, &c);
    dst[2 * j]     = xe * c - xo * s;
    dst[2 * j + 1] = xe * s + xo * c;
  }
}

__device__ __forceinline__ v16bf pack_bf16(const float* __restrict__ t) {
  v16bf r;
#pragma unroll
  for (int e = 0; e < 16; ++e) r[e] = (__bf16)t[e];
  return r;
}

// A fragment (16x32 bf16, M x K): lane<16 -> elems 0-7 = K base..base+7,
// elems 8-15 = K base+16..23, base = 32*chunk + 8*half.
__device__ __forceinline__ v16bf frag_a(const float* __restrict__ row, float pos,
                                        int chunk, int half) {
  float t[16];
  const int base = 32 * chunk + 8 * half;
  rope8(row, pos, base, t);
  rope8(row, pos, base + 16, t + 8);
  return pack_bf16(t);
}

// B fragment (32x16 bf16, K x N): elems 0-7 = K base..base+7, elems 8-15 = K base+16..23, base = 32*chunk + 8*half.
__device__ __forceinline__ v16bf frag_b(const float* __restrict__ row, float pos,
                                        int chunk, int half) {
  float t[16];
  const int base = 32 * chunk + 8 * half;
  rope8(row, pos, base, t);
  rope8(row, pos, base + 16, t + 8);
  return pack_bf16(t);
}

// ---------- shared top-k / softmax / AV machinery ----------

struct TopK {
  float tv[2][2];   // rows 2w, 2w+1; 2 slots per lane -> 64 entries per row
  int   ti[2][2];
  float rmin[2];    // wave-uniform running min + slot location
  int   rloc[2];
};

__device__ __forceinline__ void topk_init(TopK& st) {
#pragma unroll
  for (int rr = 0; rr < 2; ++rr) {
    st.tv[rr][0] = st.tv[rr][1] = kNegInf;
    st.ti[rr][0] = st.ti[rr][1] = 0;
    st.rmin[rr] = kNegInf;
    st.rloc[rr] = 0;
  }
}

__device__ __forceinline__ void topk_merge(TopK& st,
                                           const float (*cands)[kTileM][16],
                                           int wave, int lane, int g) {
#pragma unroll
  for (int rr = 0; rr < 2; ++rr) {
    const int row = 2 * wave + rr;
    // candidate (s*32 + j) lives at cands[2*s + j/16][row][j%16] for j = lane
    float cd[4];
#pragma unroll
    for (int s = 0; s < 4; ++s)
      cd[s] = cands[2 * s + (lane >> 4)][row][lane & 15];

#pragma unroll
    for (int s = 0; s < 4; ++s) {
      // wave32 ballot: only survivors of the running-min threshold get touched
      unsigned m = __builtin_amdgcn_ballot_w32(cd[s] > st.rmin[rr]);
      while (m) {
        const int j = __builtin_ctz(m);
        m &= m - 1;
        const float cv = __shfl(cd[s], j, 32);       // wave-uniform
        if (cv > st.rmin[rr]) {                      // min may have risen
          const int key = g * 128 + s * 32 + j;
          const int ml  = st.rloc[rr];
          if (lane == (ml >> 1)) {
            if (ml & 1) { st.tv[rr][1] = cv; st.ti[rr][1] = key; }
            else        { st.tv[rr][0] = cv; st.ti[rr][0] = key; }
          }
          float mv; int loc;
          if (st.tv[rr][0] <= st.tv[rr][1]) { mv = st.tv[rr][0]; loc = lane * 2; }
          else                              { mv = st.tv[rr][1]; loc = lane * 2 + 1; }
#pragma unroll
          for (int off = 16; off > 0; off >>= 1) {
            const float ov = __shfl_xor(mv, off, 32);
            const int   ol = __shfl_xor(loc, off, 32);
            if (ov < mv) { mv = ov; loc = ol; }
          }
          st.rmin[rr] = mv;
          st.rloc[rr] = loc;
        }
      }
    }
  }
}

__device__ __forceinline__ void topk_finalize(const TopK& st,
                                              const float* __restrict__ vb,
                                              float* __restrict__ out,
                                              int b, int t0, int wave, int lane) {
#pragma unroll
  for (int rr = 0; rr < 2; ++rr) {
    const int t = t0 + 2 * wave + rr;

    float m = fmaxf(st.tv[rr][0], st.tv[rr][1]);
#pragma unroll
    for (int off = 16; off > 0; off >>= 1)
      m = fmaxf(m, __shfl_xor(m, off, 32));

    const float e0 = __expf(st.tv[rr][0] - m);
    const float e1 = __expf(st.tv[rr][1] - m);
    float s = e0 + e1;
#pragma unroll
    for (int off = 16; off > 0; off >>= 1)
      s += __shfl_xor(s, off, 32);
    const float inv = 1.0f / s;
    const float w0 = e0 * inv;
    const float w1 = e1 * inv;

    float4 acc = make_float4(0.f, 0.f, 0.f, 0.f);
    for (int e = 0; e < 64; ++e) {
      const int   sl   = e >> 5;
      const float wsrc = sl ? w1 : w0;
      const int   isrc = sl ? st.ti[rr][1] : st.ti[rr][0];
      const float wj  = __shfl(wsrc, e & 31, 32);
      const int   idx = __shfl(isrc, e & 31, 32);
      if (wj > 0.f) {                                // uniform branch
        const float4 vv = *(const float4*)(vb + (size_t)idx * kDim + 4 * lane);
        acc.x += wj * vv.x;
        acc.y += wj * vv.y;
        acc.z += wj * vv.z;
        acc.w += wj * vv.w;
      }
    }
    *(float4*)(out + ((size_t)b * kSeq + t) * kDim + 4 * lane) = acc;
  }
}

// ---------- phase 1: RoPE once + pack bf16 WMMA fragments ----------
// One block = one 16-row tile; tid = c*32 + half*16 + srow so that the block
// writes a fully linear, coalesced 4KB tile: entry index within tile == tid.
__global__ __launch_bounds__(128)
void rope_pack_kernel(const float* __restrict__ q, const float* __restrict__ k,
                      unsigned short* __restrict__ qpack,
                      unsigned short* __restrict__ kpack) {
  const int tid  = threadIdx.x;       // 0..127
  const int srow = tid & 15;
  const int half = (tid >> 4) & 1;
  const int c    = tid >> 5;
  const int kt   = blockIdx.x;
  const int b    = blockIdx.y;
  const int row  = kt * kTileM + srow;
  const size_t outIdx = (((size_t)b * kTiles + kt) * 128 + tid) * 16;

  if (blockIdx.z == 0) {  // Q -> A-matrix layout
    const float* src = q + ((size_t)b * kSeq + row) * kDim;
    *(v16bf*)(qpack + outIdx) = frag_a(src, (float)row, c, half);
  } else {                // K -> B-matrix layout
    const float* src = k + ((size_t)b * kSeq + row) * kDim;
    *(v16bf*)(kpack + outIdx) = frag_b(src, (float)row, c, half);
  }
}

// ---------- phase 2: WMMA logits + streaming top-64 + softmax-V ----------
__global__ __launch_bounds__(256, 2)
void topk_attn_packed_kernel(const unsigned short* __restrict__ qpack,
                             const unsigned short* __restrict__ kpack,
                             const float* __restrict__ v,
                             float* __restrict__ out) {
  const int tile = blockIdx.x;
  const int b    = blockIdx.y;
  const int t0   = tile * kTileM;
  const int lane = threadIdx.x & 31;
  const int wave = threadIdx.x >> 5;
  const int half = lane >> 4;
  const int lrow = lane & 15;

  __shared__ float cands[kWaves][kTileM][16];
  const float* __restrict__ vb = v + (size_t)b * kSeq * kDim;

  // Q fragments: direct coalesced b128 loads of pre-packed bf16
  v16bf qf[4];
  const unsigned short* qbase = qpack + ((size_t)b * kTiles + tile) * 2048;
#pragma unroll
  for (int c = 0; c < 4; ++c)
    qf[c] = *(const v16bf*)(qbase + (c * 32 + lane) * 16);

  TopK st;
  topk_init(st);

  const int ntiles  = tile + 1;
  const int ngroups = (ntiles + kWaves - 1) / kWaves;
  const unsigned short* kbat = kpack + (size_t)b * kTiles * 2048;

  // software-pipelined K fragment buffer (registers)
  v16bf kf[4];
  if (wave < ntiles) {
#pragma unroll
    for (int c = 0; c < 4; ++c)
      kf[c] = *(const v16bf*)(kbat + (size_t)wave * 2048 + (c * 32 + lane) * 16);
  }

  for (int g = 0; g < ngroups; ++g) {
    const int kt = g * kWaves + wave;
    if (kt < ntiles) {
      // 4 back-to-back WMMAs on the pre-loaded fragments
      v8f acc = {};
#pragma unroll
      for (int c = 0; c < 4; ++c)
        acc = __builtin_amdgcn_wmma_f32_16x16x32_bf16(
            false, qf[c], false, kf[c], (short)0, acc, false, false);

      const int s0 = kt * 16;
#pragma unroll
      for (int r = 0; r < 8; ++r) {
        const int row = r + 8 * half;
        const float val = (s0 + lrow <= t0 + row) ? acc[r] * kScale : kNegInf;
        cands[wave][row][lrow] = val;
      }
    } else {
#pragma unroll
      for (int r = 0; r < 8; ++r)
        cands[wave][r + 8 * half][lrow] = kNegInf;
    }

    // Issue next group's fragment loads NOW: the barrier + top-k merge below
    // (VALU/shfl/DS work) hides the entire global->VGPR latency.
    const int ktn = kt + kWaves;
    if (ktn < ntiles) {
      {  // prefetch two groups ahead (4KB tile, 128B per lane)
        int pkt = ktn + kWaves;
        if (pkt >= kTiles) pkt = ktn;
        __builtin_prefetch(kbat + (size_t)pkt * 2048 + (size_t)lane * 64, 0, 1);
      }
#pragma unroll
      for (int c = 0; c < 4; ++c)
        kf[c] = *(const v16bf*)(kbat + (size_t)ktn * 2048 + (c * 32 + lane) * 16);
    }

    __syncthreads();
    topk_merge(st, cands, wave, lane, g);
    __syncthreads();
  }

  topk_finalize(st, vb, out, b, t0, wave, lane);
}

// ---------- fallback: fused RoPE-in-loop version (no workspace needed) ----------
__global__ __launch_bounds__(256, 2)
void topk_attn_fused_kernel(const float* __restrict__ q,
                            const float* __restrict__ k,
                            const float* __restrict__ v,
                            float* __restrict__ out) {
  const int tile = blockIdx.x;
  const int b    = blockIdx.y;
  const int t0   = tile * kTileM;
  const int lane = threadIdx.x & 31;
  const int wave = threadIdx.x >> 5;
  const int half = lane >> 4;
  const int lrow = lane & 15;

  __shared__ float cands[kWaves][kTileM][16];

  const float* __restrict__ qb = q + (size_t)b * kSeq * kDim;
  const float* __restrict__ kb = k + (size_t)b * kSeq * kDim;
  const float* __restrict__ vb = v + (size_t)b * kSeq * kDim;

  const int qrow = t0 + lrow;
  v16bf qf[4];
#pragma unroll
  for (int c = 0; c < 4; ++c)
    qf[c] = frag_a(qb + (size_t)qrow * kDim, (float)qrow, c, half);

  TopK st;
  topk_init(st);

  const int ntiles  = tile + 1;
  const int ngroups = (ntiles + kWaves - 1) / kWaves;

  for (int g = 0; g < ngroups; ++g) {
    const int kt = g * kWaves + wave;
    if (kt < ntiles) {
      const int s0   = kt * 16;
      const int krow = s0 + lrow;
      const float* __restrict__ kr = kb + (size_t)krow * kDim;
      v8f acc = {};
#pragma unroll
      for (int c = 0; c < 4; ++c) {
        const v16bf kf = frag_b(kr, (float)krow, c, half);
        acc = __builtin_amdgcn_wmma_f32_16x16x32_bf16(
            false, qf[c], false, kf, (short)0, acc, false, false);
      }
#pragma unroll
      for (int r = 0; r < 8; ++r) {
        const int row = r + 8 * half;
        const float val = (s0 + lrow <= t0 + row) ? acc[r] * kScale : kNegInf;
        cands[wave][row][lrow] = val;
      }
    } else {
#pragma unroll
      for (int r = 0; r < 8; ++r)
        cands[wave][r + 8 * half][lrow] = kNegInf;
    }
    __syncthreads();
    topk_merge(st, cands, wave, lane, g);
    __syncthreads();
  }

  topk_finalize(st, vb, out, b, t0, wave, lane);
}

} // namespace

extern "C" void kernel_launch(void* const* d_in, const int* in_sizes, int n_in,
                              void* d_out, int out_size, void* d_ws, size_t ws_size,
                              hipStream_t stream) {
  (void)in_sizes; (void)n_in; (void)out_size;
  const float* q = (const float*)d_in[0];
  const float* k = (const float*)d_in[1];
  const float* v = (const float*)d_in[2];
  float* out = (float*)d_out;

  // packed bf16 fragments: 4 MB for Q + 4 MB for K
  const size_t packHalf = (size_t)kBatch * kTiles * 128 * 16;  // ushort elements
  const size_t need = 2 * packHalf * sizeof(unsigned short);

  if (ws_size >= need) {
    unsigned short* qpack = (unsigned short*)d_ws;
    unsigned short* kpack = qpack + packHalf;
    hipLaunchKernelGGL(rope_pack_kernel, dim3(kTiles, kBatch, 2), dim3(128), 0,
                       stream, q, k, qpack, kpack);
    hipLaunchKernelGGL(topk_attn_packed_kernel, dim3(kTiles, kBatch), dim3(256),
                       0, stream, qpack, kpack, v, out);
  } else {
    hipLaunchKernelGGL(topk_attn_fused_kernel, dim3(kTiles, kBatch), dim3(256),
                       0, stream, q, k, v, out);
  }
}
